// HandNet_38354057953347
// MI455X (gfx1250) — compile-verified, matrix operand order checked by reading, not run on hardware
//
#include <hip/hip_runtime.h>
#include <hip/hip_bf16.h>
#include <math.h>

// ---------------- problem constants (from reference) ----------------
static constexpr int G      = 8192;
static constexpr int JS     = 17;
static constexpr int JT     = 13;
static constexpr int NL     = G * JS;        // 139264 nodes per encoder side
static constexpr int NSRC   = 2 * NL;        // 278528 encoder nodes
static constexpr int ES     = 2 * (JS - 1) * G; // 262144 encoder edges per side
static constexpr int HALF_T = G * JT;        // 106496 decoder nodes per side
static constexpr int NTGT   = 2 * HALF_T;    // 212992 decoder nodes
static constexpr int ET     = 2 * (JT - 1) * G; // 196608 decoder edges per side
static constexpr int MROWS  = 2 * G;         // 16384 GEMM rows
static constexpr int KDIM   = JS * 64;       // 1088 GEMM K
static constexpr int NCOLS  = JT * 64;       // 832 GEMM N
static constexpr int XDW    = 66;            // decoder input width (64 + lower + upper)
static constexpr int KSTEPS = KDIM / 32;     // 34 WMMA K-steps

// d_out layout: ang_l | pos_l | ang_r | pos_r
static constexpr int OUT_ANG_L = 0;
static constexpr int OUT_POS_L = HALF_T;                 // 106496
static constexpr int OUT_ANG_R = HALF_T + 3 * HALF_T;    // 425984
static constexpr int OUT_POS_R = OUT_ANG_R + HALF_T;     // 532480

// workspace offsets in float units (regions are reused across phases)
static constexpr size_t OFF_A   = 0;                      // h1 -> hf16 -> d1   (8,912,896 f32 slots)
static constexpr size_t OFF_B   = 8912896;                // x0 -> h2 -> d2     (8,912,896)
static constexpr size_t OFF_C   = 17825792;               // h3 -> zx[NTGT,66]  (17,825,792)
static constexpr size_t OFF_W16 = 35651584;               // trans_W in f16     (452,608 f32 slots)
static constexpr size_t OFF_ANG = 36104192;               // dec3 raw -> angles (212,992)
// total: 36,317,184 floats = ~145.3 MB of d_ws

typedef _Float16 v16h __attribute__((ext_vector_type(16)));
typedef _Float16 v8h  __attribute__((ext_vector_type(8)));
typedef float    v8f  __attribute__((ext_vector_type(8)));
typedef unsigned int v4u __attribute__((ext_vector_type(4)));
typedef unsigned int v8u __attribute__((ext_vector_type(8)));

// ---------------- small utility kernels ----------------

__global__ void concat_x_kernel(const float* __restrict__ l, const float* __restrict__ r,
                                float* __restrict__ x0) {
    int idx = blockIdx.x * blockDim.x + threadIdx.x;
    int total = NSRC * 3;
    if (idx >= total) return;
    x0[idx] = (idx < NL * 3) ? l[idx] : r[idx - NL * 3];
}

__global__ void f32_to_f16_kernel(const float* __restrict__ src, _Float16* __restrict__ dst, int n) {
    int i = blockIdx.x * blockDim.x + threadIdx.x;
    if (i < n) dst[i] = (_Float16)src[i];
}

__global__ void fill_limits_kernel(float* __restrict__ zx,
                                   const float* __restrict__ lower,
                                   const float* __restrict__ upper) {
    int n = blockIdx.x * blockDim.x + threadIdx.x;
    if (n >= NTGT) return;
    int nb = (n < HALF_T) ? n : n - HALF_T;
    zx[(size_t)n * XDW + 64] = lower[nb];
    zx[(size_t)n * XDW + 65] = upper[nb];
}

// ---------------- SpatialBasicBlock: node residual path ----------------
// out[n,o] = bu[o] + sum_i Wu[o,i] * xin[n,i]     (agg added later by atomics)
__global__ void node_up_kernel(const float* __restrict__ xin,
                               const float* __restrict__ Wu, const float* __restrict__ bu,
                               float* __restrict__ out,
                               int in_dim, int out_dim, int total) {
    int idx = blockIdx.x * blockDim.x + threadIdx.x;
    if (idx >= total) return;
    int n = idx / out_dim, o = idx % out_dim;
    const float* x = xin + (size_t)n * in_dim;
    const float* w = Wu + (size_t)o * in_dim;
    float s = bu[o];
    for (int i = 0; i < in_dim; ++i) s += w[i] * x[i];
    out[(size_t)n * out_dim + o] = s;
}

// ---------------- SpatialBasicBlock: edge message + scatter-add ----------------
// m = leaky_relu(Wl @ [x[dst], x[src], ea] + bl); out[dst] += m (atomic)
__global__ void edge_block_kernel(const float* __restrict__ xin,
                                  const int* __restrict__ ei_a, const float* __restrict__ ea_a,
                                  const int* __restrict__ ei_b, const float* __restrict__ ea_b,
                                  int E_side, int node_off_b,
                                  const float* __restrict__ Wl, const float* __restrict__ bl,
                                  float* __restrict__ out,
                                  int in_dim, int edge_dim, int out_dim, int total) {
    int idx = blockIdx.x * blockDim.x + threadIdx.x;
    if (idx >= total) return;
    int e = idx / out_dim, o = idx % out_dim;
    int src, dst;
    const float* ea;
    if (e < E_side) {
        src = ei_a[e];
        dst = ei_a[E_side + e];
        ea  = ea_a + (size_t)e * edge_dim;
    } else {
        int e2 = e - E_side;
        src = ei_b[e2] + node_off_b;
        dst = ei_b[E_side + e2] + node_off_b;
        ea  = ea_b + (size_t)e2 * edge_dim;
    }
    int zlen = 2 * in_dim + edge_dim;
    const float* w  = Wl + (size_t)o * zlen;
    const float* xd = xin + (size_t)dst * in_dim;
    const float* xs = xin + (size_t)src * in_dim;
    float s = bl[o];
    for (int i = 0; i < in_dim; ++i)  s += w[i] * xd[i];
    for (int i = 0; i < in_dim; ++i)  s += w[in_dim + i] * xs[i];
    for (int j = 0; j < edge_dim; ++j) s += w[2 * in_dim + j] * ea[j];
    s = (s > 0.0f) ? s : 0.01f * s;   // leaky_relu, default slope
    atomicAdd(out + (size_t)dst * out_dim + o, s);
}

// ---------------- TDM: stage a 64(n) x 32(k) f16 B-slab into LDS ----------------
// 2-group D#: data_size=2B, tile_dim0=32 (K contiguous), tile_dim1=64 (W rows),
// tensor_dim0_stride = KDIM. Issued by one wave; tracked with TENSORcnt.
// NOTE: the LDS destination pointer is passed INTO the asm (both as the descriptor
// lds_addr and as a dummy operand) so the shared array escapes -> the compiler
// must keep the ds_load consumers instead of folding them to undef.
__device__ __forceinline__ void tdm_load_b_slab(const _Float16* gtile, _Float16* ldst) {
    unsigned long long ga = (unsigned long long)(uintptr_t)gtile;
    unsigned lds_off = (unsigned)(uintptr_t)ldst;   // low 32b of flat LDS addr = LDS byte offset
    v4u g0;
    g0[0] = 1u;                                            // count=1, user descriptor
    g0[1] = lds_off;                                       // lds_addr (bytes)
    g0[2] = (unsigned)(ga & 0xffffffffull);                // global_addr[31:0]
    g0[3] = (unsigned)((ga >> 32) & 0x1ffffffull) | (2u << 30); // global_addr[56:32] | type=2
    v8u g1;
    g1[0] = 1u << 16;                                      // data_size=1 (2 bytes)
    g1[1] = ((unsigned)KDIM & 0xffffu) << 16;              // tensor_dim0[15:0] (bits 79:48)
    g1[2] = ((unsigned)KDIM >> 16) | (((unsigned)NCOLS & 0xffffu) << 16); // td0 hi | td1 lo
    g1[3] = ((unsigned)NCOLS >> 16) | (32u << 16);         // td1 hi | tile_dim0=32
    g1[4] = 64u;                                           // tile_dim1=64, tile_dim2=0
    g1[5] = (unsigned)KDIM;                                // tensor_dim0_stride lo
    g1[6] = 0u;                                            // stride0 hi | stride1 lo
    g1[7] = 0u;                                            // stride1 hi
    asm volatile("tensor_load_to_lds %0, %1"
                 :: "s"(g0), "s"(g1), "s"(ldst)            // %2 unused: escapes LDS pointer
                 : "memory");
}

// ---------------- transform GEMM with WMMA + TDM double-buffered B ----------------
// zx[node, f] = tanh( A[m,:] . W[n,:] + bias[n] ),  node = m*13 + n/64, f = n%64
// Block: 8 waves -> 128 rows x 64 cols.  Wave: 16 rows x 4 N-tiles (A reused 4x).
// B slab for the whole block comes from LDS, filled by the Tensor Data Mover.
__global__ __launch_bounds__(256) void trans_gemm_kernel(
        const _Float16* __restrict__ A,
        const _Float16* __restrict__ W,
        const float* __restrict__ bias,
        float* __restrict__ zx) {
    __shared__ __align__(16) _Float16 Bsl[2][64 * 32];     // 2 x 4KB double buffer

    const int lane = threadIdx.x & 31;
    const int wave = threadIdx.x >> 5;
    const int mt   = blockIdx.y * 8 + wave;                // 0..1023 M-tile
    const int nb0  = blockIdx.x * 64;                      // column block start
    const int ml   = lane & 15;
    const int khalf = (lane >> 4) * 8;                     // A/B 16-bit fragment K sub-offset

    const _Float16* arow = A + (size_t)(mt * 16 + ml) * KDIM + khalf;

    v8f acc[4] = {{}, {}, {}, {}};

    // prologue: stage B slab for K-step 0 into buffer 0
    if (wave == 0) {
        tdm_load_b_slab(W + (size_t)nb0 * KDIM, &Bsl[0][0]);
    }

    for (int s = 0; s < KSTEPS; ++s) {
        const int buf = s & 1;
        if (wave == 0) __builtin_amdgcn_s_wait_tensorcnt(0);
        __syncthreads();                                   // B slab for step s visible
        if (wave == 0 && (s + 1) < KSTEPS) {               // prefetch next slab (other buffer)
            tdm_load_b_slab(W + (size_t)nb0 * KDIM + (size_t)(s + 1) * 32,
                            &Bsl[buf ^ 1][0]);
        }
        if ((s + 4) < KSTEPS)
            __builtin_prefetch(arow + (size_t)(s + 4) * 32, 0, 3);  // global_prefetch_b8 (near)

        v8h alo = *(const v8h*)(arow + (size_t)s * 32);
        v8h ahi = *(const v8h*)(arow + (size_t)s * 32 + 16);
        v16h a = __builtin_shufflevector(alo, ahi, 0, 1, 2, 3, 4, 5, 6, 7,
                                         8, 9, 10, 11, 12, 13, 14, 15);
#pragma unroll
        for (int t = 0; t < 4; ++t) {
            const _Float16* bp = &Bsl[buf][(t * 16 + ml) * 32 + khalf];
            v8h blo = *(const v8h*)bp;
            v8h bhi = *(const v8h*)(bp + 16);
            v16h b = __builtin_shufflevector(blo, bhi, 0, 1, 2, 3, 4, 5, 6, 7,
                                             8, 9, 10, 11, 12, 13, 14, 15);
            acc[t] = __builtin_amdgcn_wmma_f32_16x16x32_f16(
                    /*neg_a=*/false, a, /*neg_b=*/false, b,
                    /*c_mod=*/(short)0, acc[t], /*reuse_a=*/false, /*reuse_b=*/false);
        }
    }

    // epilogue: bias + tanh, scatter into decoder xd layout
    const int mhi = (lane >> 4) ? 8 : 0;                   // C/D: VGPR r -> row r / r+8
#pragma unroll
    for (int t = 0; t < 4; ++t) {
        int ncol = nb0 + t * 16 + ml;                      // C/D: lane&15 is the N index
        float bv = bias[ncol];
#pragma unroll
        for (int r = 0; r < 8; ++r) {
            int mrow = mt * 16 + r + mhi;
            float v = tanhf(acc[t][r] + bv);
            int node = mrow * 13 + (ncol >> 6);
            int f = ncol & 63;
            zx[(size_t)node * XDW + f] = v;
        }
    }
}

// ---------------- angle limits + output copy ----------------
__global__ void angle_kernel(float* __restrict__ araw,
                             const float* __restrict__ lower, const float* __restrict__ upper,
                             float* __restrict__ dout) {
    int n = blockIdx.x * blockDim.x + threadIdx.x;
    if (n >= NTGT) return;
    int nb = (n < HALF_T) ? n : n - HALF_T;
    float lo = lower[nb], up = upper[nb];
    float a = tanhf(araw[n]);
    float ang = lo + (up - lo) * (a + 1.0f) * 0.5f;
    araw[n] = ang;
    if (n < HALF_T) dout[OUT_ANG_L + n] = ang;
    else            dout[OUT_ANG_R + (n - HALF_T)] = ang;
}

// ---------------- forward kinematics (one thread per graph) ----------------
__global__ void fk_kernel(const float* __restrict__ ang,
                          const int* __restrict__ parent,
                          const float* __restrict__ offset,
                          const float* __restrict__ axis,
                          float* __restrict__ dout) {
    int g = blockIdx.x * blockDim.x + threadIdx.x;
    if (g >= MROWS) return;
    int gb = g & (G - 1);   // offset/axis replicate across the two sides

    float rot[JT][9];
    float pos[JT][3];
    for (int j = 0; j < JT; ++j) {
        int p = parent[j];  // topology identical for every graph
        float a = ang[(size_t)g * JT + j];
        const float* off = offset + ((size_t)gb * JT + j) * 3;
        const float* ax  = axis   + ((size_t)gb * JT + j) * 3;
        float c = cosf(a), s = sinf(a), oc = 1.0f - c;
        float x = ax[0], y = ax[1], z = ax[2];
        float Rl[9] = { c + oc * x * x, oc * x * y - s * z, oc * x * z + s * y,
                        oc * y * x + s * z, c + oc * y * y, oc * y * z - s * x,
                        oc * z * x - s * y, oc * z * y + s * x, c + oc * z * z };
        float Rp[9], Pp[3];
        if (p < 0) {
            Rp[0] = 1; Rp[1] = 0; Rp[2] = 0;
            Rp[3] = 0; Rp[4] = 1; Rp[5] = 0;
            Rp[6] = 0; Rp[7] = 0; Rp[8] = 1;
            Pp[0] = Pp[1] = Pp[2] = 0.0f;
        } else {
            for (int k = 0; k < 9; ++k) Rp[k] = rot[p][k];
            for (int k = 0; k < 3; ++k) Pp[k] = pos[p][k];
        }
        for (int rI = 0; rI < 3; ++rI) {
            pos[j][rI] = Pp[rI] + Rp[rI * 3 + 0] * off[0] + Rp[rI * 3 + 1] * off[1] + Rp[rI * 3 + 2] * off[2];
            for (int cI = 0; cI < 3; ++cI) {
                rot[j][rI * 3 + cI] = Rp[rI * 3 + 0] * Rl[0 * 3 + cI]
                                    + Rp[rI * 3 + 1] * Rl[1 * 3 + cI]
                                    + Rp[rI * 3 + 2] * Rl[2 * 3 + cI];
            }
        }
    }
    size_t base = (g < G) ? (OUT_POS_L + (size_t)g * JT * 3)
                          : (OUT_POS_R + (size_t)(g - G) * JT * 3);
    for (int j = 0; j < JT; ++j)
        for (int cI = 0; cI < 3; ++cI)
            dout[base + j * 3 + cI] = pos[j][cI];
}

// ---------------- launch ----------------
static inline unsigned nblk(long long n, int t) { return (unsigned)((n + t - 1) / t); }

extern "C" void kernel_launch(void* const* d_in, const int* in_sizes, int n_in,
                              void* d_out, int out_size, void* d_ws, size_t ws_size,
                              hipStream_t stream) {
    (void)in_sizes; (void)n_in; (void)out_size; (void)ws_size;
    const float* l_x   = (const float*)d_in[0];
    const float* r_x   = (const float*)d_in[1];
    const int*   l_ei  = (const int*)d_in[2];
    const int*   r_ei  = (const int*)d_in[3];
    const float* l_ea  = (const float*)d_in[4];
    const float* r_ea  = (const float*)d_in[5];
    const int*   t_ei  = (const int*)d_in[6];
    const float* t_ea  = (const float*)d_in[7];
    const float* lower = (const float*)d_in[8];
    const float* upper = (const float*)d_in[9];
    const float* offs  = (const float*)d_in[10];
    const float* axis  = (const float*)d_in[11];
    const int*   par   = (const int*)d_in[12];
    // d_in[13] = num_graphs (compile-time constant here)
    const float* e1lW = (const float*)d_in[14]; const float* e1lB = (const float*)d_in[15];
    const float* e1uW = (const float*)d_in[16]; const float* e1uB = (const float*)d_in[17];
    const float* e2lW = (const float*)d_in[18]; const float* e2lB = (const float*)d_in[19];
    const float* e2uW = (const float*)d_in[20]; const float* e2uB = (const float*)d_in[21];
    const float* e3lW = (const float*)d_in[22]; const float* e3lB = (const float*)d_in[23];
    const float* e3uW = (const float*)d_in[24]; const float* e3uB = (const float*)d_in[25];
    const float* trW  = (const float*)d_in[26]; const float* trB  = (const float*)d_in[27];
    const float* d1lW = (const float*)d_in[28]; const float* d1lB = (const float*)d_in[29];
    const float* d1uW = (const float*)d_in[30]; const float* d1uB = (const float*)d_in[31];
    const float* d2lW = (const float*)d_in[32]; const float* d2lB = (const float*)d_in[33];
    const float* d2uW = (const float*)d_in[34]; const float* d2uB = (const float*)d_in[35];
    const float* d3lW = (const float*)d_in[36]; const float* d3lB = (const float*)d_in[37];
    const float* d3uW = (const float*)d_in[38]; const float* d3uB = (const float*)d_in[39];

    float* wsf = (float*)d_ws;
    float* regA = wsf + OFF_A;              // h1 -> (hf16) -> d1
    float* regB = wsf + OFF_B;              // x0 -> h2 -> d2
    float* regC = wsf + OFF_C;              // h3 -> zx
    _Float16* hf16 = (_Float16*)regA;
    _Float16* wf16 = (_Float16*)(wsf + OFF_W16);
    float* araw = wsf + OFF_ANG;
    float* outp = (float*)d_out;

    const int T = 256;

    // -------- encoder --------
    concat_x_kernel<<<nblk((long long)NSRC * 3, T), T, 0, stream>>>(l_x, r_x, regB);

    // enc1: 3 -> 16
    node_up_kernel<<<nblk((long long)NSRC * 16, T), T, 0, stream>>>(
        regB, e1uW, e1uB, regA, 3, 16, NSRC * 16);
    edge_block_kernel<<<nblk((long long)2 * ES * 16, T), T, 0, stream>>>(
        regB, l_ei, l_ea, r_ei, r_ea, ES, NL, e1lW, e1lB, regA, 3, 3, 16, 2 * ES * 16);

    // enc2: 16 -> 32   (h2 overwrites x0; x0 no longer needed)
    node_up_kernel<<<nblk((long long)NSRC * 32, T), T, 0, stream>>>(
        regA, e2uW, e2uB, regB, 16, 32, NSRC * 32);
    edge_block_kernel<<<nblk((long long)2 * ES * 32, T), T, 0, stream>>>(
        regA, l_ei, l_ea, r_ei, r_ea, ES, NL, e2lW, e2lB, regB, 16, 3, 32, 2 * ES * 32);

    // enc3: 32 -> 64
    node_up_kernel<<<nblk((long long)NSRC * 64, T), T, 0, stream>>>(
        regB, e3uW, e3uB, regC, 32, 64, NSRC * 64);
    edge_block_kernel<<<nblk((long long)2 * ES * 64, T), T, 0, stream>>>(
        regB, l_ei, l_ea, r_ei, r_ea, ES, NL, e3lW, e3lB, regC, 32, 3, 64, 2 * ES * 64);

    // -------- transform (WMMA GEMM, TDM-staged B) --------
    f32_to_f16_kernel<<<nblk((long long)MROWS * KDIM, T), T, 0, stream>>>(regC, hf16, MROWS * KDIM);
    f32_to_f16_kernel<<<nblk((long long)NCOLS * KDIM, T), T, 0, stream>>>(trW, wf16, NCOLS * KDIM);

    // zx gets columns 64/65 = joint limits; GEMM fills columns 0..63 (tanh fused)
    fill_limits_kernel<<<nblk(NTGT, T), T, 0, stream>>>(regC, lower, upper);
    dim3 ggrid(NCOLS / 64, MROWS / 16 / 8);  // (13, 128), 8 waves per block
    trans_gemm_kernel<<<ggrid, 256, 0, stream>>>(hf16, wf16, trB, regC);

    // -------- decoder --------
    // dec1: 66 -> 32  (d1 overwrites hf16; dead after GEMM)
    node_up_kernel<<<nblk((long long)NTGT * 32, T), T, 0, stream>>>(
        regC, d1uW, d1uB, regA, XDW, 32, NTGT * 32);
    edge_block_kernel<<<nblk((long long)2 * ET * 32, T), T, 0, stream>>>(
        regC, t_ei, t_ea, t_ei, t_ea, ET, HALF_T, d1lW, d1lB, regA, XDW, 6, 32, 2 * ET * 32);

    // dec2: 32 -> 16
    node_up_kernel<<<nblk((long long)NTGT * 16, T), T, 0, stream>>>(
        regA, d2uW, d2uB, regB, 32, 16, NTGT * 16);
    edge_block_kernel<<<nblk((long long)2 * ET * 16, T), T, 0, stream>>>(
        regA, t_ei, t_ea, t_ei, t_ea, ET, HALF_T, d2lW, d2lB, regB, 32, 6, 16, 2 * ET * 16);

    // dec3: 16 -> 1
    node_up_kernel<<<nblk(NTGT, T), T, 0, stream>>>(
        regB, d3uW, d3uB, araw, 16, 1, NTGT);
    edge_block_kernel<<<nblk((long long)2 * ET, T), T, 0, stream>>>(
        regB, t_ei, t_ea, t_ei, t_ea, ET, HALF_T, d3lW, d3lB, araw, 16, 6, 1, 2 * ET);

    // tanh + joint-limit mapping; write ang halves into d_out
    angle_kernel<<<nblk(NTGT, T), T, 0, stream>>>(araw, lower, upper, outp);

    // forward kinematics; write pos halves into d_out
    fk_kernel<<<nblk(MROWS, T), T, 0, stream>>>(araw, par, offs, axis, outp);
}
